// SpearmanLoss_63093069578904
// MI455X (gfx1250) — compile-verified, matrix operand order checked by reading, not run on hardware
//
#include <hip/hip_runtime.h>
#include <hip/hip_bf16.h>
#include <stdint.h>

typedef __attribute__((ext_vector_type(16))) _Float16 v16h;
typedef __attribute__((ext_vector_type(8)))  float    v8f;
typedef __attribute__((ext_vector_type(4)))  uint32_t u32x4;
typedef __attribute__((ext_vector_type(8)))  uint32_t u32x8;

#define N_MAX 16384

// sigma(t) = 0.5 + 0.5*tanh(t/2), t = (xi-xj)/0.01  =>  tanh arg = (xi-xj)*50
#define K_TANH 50.0f
// exp2 fallback scale: (1/temp)*log2(e)
#define K_EXP2 144.26950408889634f

#if __has_builtin(__builtin_amdgcn_tanhf)
#define HAVE_TANH 1
#else
#define HAVE_TANH 0
#endif

// Returns tanh(((xi - xj)/temp)/2)  ==  2*sigma((xi-xj)/temp) - 1
__device__ __forceinline__ float sig_tanh(float xj, float xiK)
{
#if HAVE_TANH
    // xiK = xi * 50
    return __builtin_amdgcn_tanhf(__builtin_fmaf(xj, -K_TANH, xiK));
#else
    // xiK = xi * K_EXP2 ; exp2((xj-xi)*K_EXP2) = e^{-t}
    float z = __builtin_fmaf(xj, K_EXP2, -xiK);
    float s = __builtin_amdgcn_rcpf(1.0f + __builtin_amdgcn_exp2f(z));
    return __builtin_fmaf(2.0f, s, -1.0f);
#endif
}

// ---------------------------------------------------------------------------
// Phase 1: soft ranks. One block = 8 waves = 128 rows. blockIdx.y selects
// predictions (0) or targets (1). Whole input vector (64 KB) is DMA'd into
// LDS by the Tensor Data Mover; each wave sweeps K in 32-chunks, computing a
// 16x32 f16 tanh tile and reducing row-sums with v_wmma_f32_16x16x32_f16
// against an all-ones B matrix. rank_i = 1 + n/2 + 0.5 * rowsum_i.
// ---------------------------------------------------------------------------
__global__ __launch_bounds__(256, 2)
void soft_ranks_kernel(const float* __restrict__ pred,
                       const float* __restrict__ targ,
                       float* __restrict__ ranks, int n)
{
    __shared__ float xs[N_MAX];

    const float* src = (blockIdx.y == 0) ? pred : targ;
    float*       dst = ranks + (size_t)blockIdx.y * (size_t)n;

    // Never executed (n > 0 always), but guarantees the optimizer cannot
    // prove xs[] is store-free and fold its loads to poison.
    if (n < 0) xs[0] = (float)n;

    // LDS byte offset of xs: generic LDS address truncates to the
    // wave-relative LDS offset (ISA 10.2). Also captures xs's address so
    // the TDM asm below is a legal writer of it.
    const uint32_t lds_off = (uint32_t)(uintptr_t)(void*)&xs[0];

    // ---- TDM: DMA the full input vector into LDS (wave 0 issues) ----------
    if (threadIdx.x < 32) {
        uint64_t ga = (uint64_t)(uintptr_t)src;
        u32x4 g0;
        g0.x = 1u;                                  // count = 1 valid D#
        g0.y = lds_off;                             // lds_addr (bytes)
        g0.z = (uint32_t)ga;                        // global_addr[31:0]
        g0.w = (uint32_t)(ga >> 32) | (2u << 30);   // global_addr[56:32] | type=2
        u32x8 g1;
        g1.s0 = 2u << 16;                           // data_size = 4 bytes
        g1.s1 = ((uint32_t)n & 0xFFFFu) << 16;      // tensor_dim0[15:0]
        g1.s2 = ((uint32_t)n >> 16) | (1u << 16);   // tensor_dim0[31:16] | tensor_dim1=1
        g1.s3 = ((uint32_t)n & 0xFFFFu) << 16;      // tile_dim0 = n (single row tile)
        g1.s4 = 1u;                                 // tile_dim1 = 1, tile_dim2 = 0
        g1.s5 = (uint32_t)n;                        // tensor_dim0_stride[31:0]
        g1.s6 = 0u;
        g1.s7 = 0u;
        asm volatile("tensor_load_to_lds %0, %1"
                     :: "s"(g0), "s"(g1), "v"((void*)&xs[0])
                     : "memory");
        __builtin_amdgcn_s_wait_tensorcnt(0);
    }
    __syncthreads();

    const int lane  = threadIdx.x & 31;
    const int wave  = threadIdx.x >> 5;
    const int m     = lane & 15;                    // my A-matrix row
    const int iBase = (blockIdx.x * 8 + wave) * 16; // 16 rows per wave

#if HAVE_TANH
    const float xiK = xs[iBase + m] * K_TANH;
#else
    const float xiK = xs[iBase + m] * K_EXP2;
#endif

    v16h bones;
    #pragma unroll
    for (int e = 0; e < 16; ++e) bones[e] = (_Float16)1.0f;

    v8f c = {};                                     // f32 row-sum accumulator

    // A-layout (16-bit 16x32): lanes 0-15 cover K {0..7,16..23},
    // lanes 16-31 cover K {8..15,24..31}; row = lane & 15.
    const int half = (lane < 16) ? 0 : 8;

    for (int k0 = 0; k0 < n; k0 += 32) {
        const float4 q0 = *reinterpret_cast<const float4*>(&xs[k0 + half]);
        const float4 q1 = *reinterpret_cast<const float4*>(&xs[k0 + half + 4]);
        const float4 q2 = *reinterpret_cast<const float4*>(&xs[k0 + half + 16]);
        const float4 q3 = *reinterpret_cast<const float4*>(&xs[k0 + half + 20]);
        const float xj[16] = { q0.x, q0.y, q0.z, q0.w,  q1.x, q1.y, q1.z, q1.w,
                               q2.x, q2.y, q2.z, q2.w,  q3.x, q3.y, q3.z, q3.w };
        v16h a;
        #pragma unroll
        for (int e = 0; e < 16; ++e) {
            a[e] = (_Float16)sig_tanh(xj[e], xiK);  // tanh value in [-1,1]
        }
        // D = A x ones + C : every column of D holds the row sums (f32 acc)
        c = __builtin_amdgcn_wmma_f32_16x16x32_f16(false, a, false, bones,
                                                   (short)0, c, false, false);
    }

    // rank_i = 1 + n/2 + 0.5 * sum_j tanh(...)
    const float rankBase = 1.0f + 0.5f * (float)n;

    // C/D layout: lane l<16 holds rows 0..7 (c[r] = (M=r, N=l));
    // lane 16+r holds rows 8..15 (c[r'] = (M=8+r', N=r)). All N equal.
    if (lane < 8) {
        dst[iBase + lane] = __builtin_fmaf(0.5f, c[lane], rankBase);
    } else if (lane >= 16 && lane < 24) {
        dst[iBase + 8 + (lane - 16)] = __builtin_fmaf(0.5f, c[lane - 16], rankBase);
    }
}

// ---------------------------------------------------------------------------
// Phase 2: Pearson correlation of the two rank vectors (double accumulators),
// single block, output = -corr.
// ---------------------------------------------------------------------------
__global__ __launch_bounds__(256)
void spearman_corr_kernel(const float* __restrict__ ranks,
                          float* __restrict__ out, int n)
{
    __shared__ double red0[256];
    __shared__ double red1[256];
    __shared__ double red2[256];
    const int tid = threadIdx.x;

    double sp = 0.0, st = 0.0;
    for (int i = tid; i < n; i += 256) { sp += ranks[i]; st += ranks[n + i]; }
    red0[tid] = sp; red1[tid] = st;
    __syncthreads();
    for (int s = 128; s > 0; s >>= 1) {
        if (tid < s) { red0[tid] += red0[tid + s]; red1[tid] += red1[tid + s]; }
        __syncthreads();
    }
    const double mp = red0[0] / (double)n;
    const double mt = red1[0] / (double)n;
    __syncthreads();

    double spp = 0.0, stt = 0.0, spt = 0.0;
    for (int i = tid; i < n; i += 256) {
        double pr = (double)ranks[i]     - mp;
        double tr = (double)ranks[n + i] - mt;
        spp += pr * pr; stt += tr * tr; spt += pr * tr;
    }
    red0[tid] = spp; red1[tid] = stt; red2[tid] = spt;
    __syncthreads();
    for (int s = 128; s > 0; s >>= 1) {
        if (tid < s) {
            red0[tid] += red0[tid + s];
            red1[tid] += red1[tid + s];
            red2[tid] += red2[tid + s];
        }
        __syncthreads();
    }
    if (tid == 0) {
        double denom = sqrt(red0[0]) * sqrt(red1[0]);
        out[0] = (float)(-(red2[0] / denom));
    }
}

extern "C" void kernel_launch(void* const* d_in, const int* in_sizes, int n_in,
                              void* d_out, int out_size, void* d_ws, size_t ws_size,
                              hipStream_t stream)
{
    const float* pred  = (const float*)d_in[0];
    const float* targ  = (const float*)d_in[1];
    float*       out   = (float*)d_out;
    float*       ranks = (float*)d_ws;          // 2*n floats of scratch
    const int    n     = in_sizes[0];           // 16384

    dim3 grid(n / 128, 2, 1);                   // 128 rows per block, 2 arrays
    soft_ranks_kernel<<<grid, 256, 0, stream>>>(pred, targ, ranks, n);
    spearman_corr_kernel<<<1, 256, 0, stream>>>(ranks, out, n);
}
